// GritTransformerLayer_59596966199892
// MI455X (gfx1250) — matrix operand order repaired and checked by reading, not compiled, not run on hardware
//
#include <hip/hip_runtime.h>
#include <hip/hip_bf16.h>
#include <math.h>

#define N_NODES 50000
#define E_EDGES 500000

typedef __attribute__((ext_vector_type(16))) _Float16 v16h;
typedef __attribute__((ext_vector_type(8)))  float    v8f;
typedef __attribute__((ext_vector_type(4)))  _Float16 h4;

// ---------------------------------------------------------------------------
// WMMA fragment helpers (wave32, V_WMMA_F32_16X16X32_F16 layouts per ISA §7.12.2)
// A (16x32 f16) comes from LDS: lane<16 holds row m=lane, K {0..7,16..23};
//   lane>=16 K {8..15,24..31} -> two contiguous 16B runs => 2x ds_load_b128.
// B comes from global in PRE-SWIZZLED fragment layout: each 32x16 K-block is
//   512 contiguous halves ordered [lane][j], so a lane's fragment is one
//   aligned 32-byte run => 2x global_load_b128 (vs 16x global_load_u16).
// C (16x16 f32): reg r -> (m = r + 8*(lane>>4), n = lane&15)
// ---------------------------------------------------------------------------
__device__ __forceinline__ v16h frag_a(const _Float16* A, int lda, int k0, int lane) {
  const int m = lane & 15, hl = lane >> 4;
  v16h a;
#pragma unroll
  for (int p = 0; p < 8; ++p) {
    const int k = k0 + ((p < 4) ? (2 * p) : (16 + 2 * (p - 4))) + 8 * hl;
    a[2 * p]     = A[m * lda + k];
    a[2 * p + 1] = A[m * lda + k + 1];
  }
  return a;
}

__device__ __forceinline__ v16h frag_b_swz(const _Float16* Wf, int blk, int lane) {
  return *(const v16h*)(Wf + (size_t)blk * 512 + lane * 16);
}

// K=64 GEMM: A from LDS (lda), B pre-swizzled (2 K-blocks per col-tile)
__device__ __forceinline__ v8f gemm_k64(const _Float16* A, int lda,
                                        const _Float16* Wf, int ct, int lane) {
  v8f c = {};
  c = __builtin_amdgcn_wmma_f32_16x16x32_f16(false, frag_a(A, lda, 0, lane),
                                             false, frag_b_swz(Wf, ct * 2 + 0, lane),
                                             (short)0, c, false, false);
  c = __builtin_amdgcn_wmma_f32_16x16x32_f16(false, frag_a(A, lda, 32, lane),
                                             false, frag_b_swz(Wf, ct * 2 + 1, lane),
                                             (short)0, c, false, false);
  return c;
}

// LayerNorm over 16 rows x 64 cols in LDS. 128 threads: 8 threads/row, shfl-xor
// reduce within groups of 8 lanes (wave32 -> groups never cross wave boundary).
__device__ __forceinline__ void ln16_rows(float (*buf)[64], const float* g, const float* bta,
                                          int t, float (*of)[64], _Float16 (*oh)[64],
                                          float* og) {
  const int r = t >> 3, l = t & 7;
  float s = 0.f, ss = 0.f;
#pragma unroll
  for (int j = 0; j < 8; ++j) { float v = buf[r][l * 8 + j]; s += v; ss += v * v; }
#pragma unroll
  for (int m = 1; m < 8; m <<= 1) { s += __shfl_xor(s, m, 8); ss += __shfl_xor(ss, m, 8); }
  const float mean = s * 0.015625f;
  const float var  = ss * 0.015625f - mean * mean;
  const float inv  = rsqrtf(var + 1e-5f);
#pragma unroll
  for (int j = 0; j < 8; ++j) {
    const int c = l * 8 + j;
    const float y = (buf[r][c] - mean) * inv * g[c] + bta[c];
    if (of) of[r][c] = y;
    if (oh) oh[r][c] = (_Float16)y;
    if (og) og[(size_t)r * 64 + c] = y;   // 8 contiguous floats/thread -> b128 stores
  }
}

// ---------------------------------------------------------------------------
// Kernel 0: pack weights to f16 in fragment-swizzled layout + pack biases.
// Fragment half index -> (K-block, lane, j) -> (k, col) of the source matrix.
// Layout (halves): Wqkv@0 (64x192,2ks) | We@12288 (64x128,2ks) | Woh@20480 |
// Woe@24576 | W1@28672 (64x128) | W2@36864 (128x64,4ks) | Wve@45056 | end 49152
// Then 192 f32 biases [bq|bk|bv].
// ---------------------------------------------------------------------------
__device__ __forceinline__ void frag_decode(int j, int Ksteps, int& k, int& col) {
  const int blk = j >> 9;
  const int w = j & 511;
  const int lane = w >> 4, jj = w & 15;
  const int hl = lane >> 4, n = lane & 15;
  const int p = jj >> 1, lo = jj & 1;
  const int ct = blk / Ksteps, ks = blk % Ksteps;
  k = ks * 32 + ((p < 4) ? (2 * p) : (16 + 2 * (p - 4))) + 8 * hl + lo;
  col = ct * 16 + n;
}

__global__ void k_prep(const float* __restrict__ Wq, const float* __restrict__ Wk,
                       const float* __restrict__ Wv, const float* __restrict__ We,
                       const float* __restrict__ Woh, const float* __restrict__ Woe,
                       const float* __restrict__ W1, const float* __restrict__ W2,
                       const float* __restrict__ VeRow,
                       const float* __restrict__ bq, const float* __restrict__ bk,
                       const float* __restrict__ bv,
                       _Float16* __restrict__ W16, float* __restrict__ bqkv) {
  const int idx = blockIdx.x * blockDim.x + threadIdx.x;
  if (idx >= 49152) {
    const int b = idx - 49152;
    if (b < 192) bqkv[b] = (b < 64) ? bq[b] : (b < 128) ? bk[b - 64] : bv[b - 128];
    return;
  }
  float v;
  int k, col;
  if (idx < 12288) {                       // Wqkv 64x192
    frag_decode(idx, 2, k, col);
    v = (col < 64) ? Wq[k * 64 + col] : (col < 128) ? Wk[k * 64 + col - 64]
                                                    : Wv[k * 64 + col - 128];
  } else if (idx < 20480) {                // We 64x128
    frag_decode(idx - 12288, 2, k, col);
    v = We[k * 128 + col];
  } else if (idx < 24576) {                // Woh 64x64
    frag_decode(idx - 20480, 2, k, col);
    v = Woh[k * 64 + col];
  } else if (idx < 28672) {                // Woe 64x64
    frag_decode(idx - 24576, 2, k, col);
    v = Woe[k * 64 + col];
  } else if (idx < 36864) {                // W1 64x128
    frag_decode(idx - 28672, 2, k, col);
    v = W1[k * 128 + col];
  } else if (idx < 45056) {                // W2 128x64 (4 K-steps)
    frag_decode(idx - 36864, 4, k, col);
    v = W2[k * 64 + col];
  } else {                                 // Wve block-diag from VeRow(8,8,8)
    frag_decode(idx - 45056, 2, k, col);
    const int h = k >> 3, d = k & 7, hn = col >> 3, cc = col & 7;
    v = (hn == h) ? VeRow[d * 64 + h * 8 + cc] : 0.f;
  }
  W16[idx] = (_Float16)v;
}

// stage a 16x64 f32 tile -> f16 LDS (float4 loads, 64-bit LDS stores)
__device__ __forceinline__ void stage_f32_to_h16(const float* __restrict__ src,
                                                 _Float16 (*dst)[64], int t) {
  for (int i = t; i < 256; i += 128) {
    const int r = i >> 4, c0 = (i & 15) * 4;
    const float4 v = *(const float4*)(src + (size_t)r * 64 + c0);
    h4 hv; hv[0] = (_Float16)v.x; hv[1] = (_Float16)v.y;
    hv[2] = (_Float16)v.z; hv[3] = (_Float16)v.w;
    *(h4*)&dst[r][c0] = hv;
  }
}

// ---------------------------------------------------------------------------
// Kernel 1: QKV = x @ [Wq|Wk|Wv] + bias   (N x 192 f32 in ws)
// ---------------------------------------------------------------------------
__global__ void k_qkv(const float* __restrict__ x, const _Float16* __restrict__ Wqkv,
                      const float* __restrict__ bqkv, float* __restrict__ qkv) {
  __shared__ _Float16 sA[16][64];
  const int t = threadIdx.x, lane = t & 31, wave = t >> 5;
  const int row0 = blockIdx.x * 16;
  stage_f32_to_h16(x + (size_t)row0 * 64, sA, t);
  __syncthreads();
  const int hl = lane >> 4, nn = lane & 15;
#pragma unroll
  for (int j = 0; j < 3; ++j) {
    const int ct = wave * 3 + j;  // 0..11
    v8f c = gemm_k64(&sA[0][0], 64, Wqkv, ct, lane);
    const int col = ct * 16 + nn;
    const float bias = bqkv[col];
#pragma unroll
    for (int r = 0; r < 8; ++r)
      qkv[(size_t)(row0 + r + 8 * hl) * 192 + col] = c[r] + bias;
  }
}

// ---------------------------------------------------------------------------
// Kernel 2: edge scores. Ee = edge_attr@We+be (WMMA in LDS);
// s = signed_sqrt((Kh[src]+Qh[dst])*Ew)+Eb -> f16 store; a = dot(s,Aw) per head,
// clip +-5; ex = exp(a-5) (softmax shift-invariance replaces segment-max);
// atomic-add ex into den[dst].
// ---------------------------------------------------------------------------
__global__ void k_edge_score(const float* __restrict__ edge_attr,
                             const _Float16* __restrict__ We16,
                             const float* __restrict__ be,
                             const float* __restrict__ qkv,
                             const int* __restrict__ src, const int* __restrict__ dst,
                             const float* __restrict__ Aw,
                             _Float16* __restrict__ s_out,
                             float* __restrict__ ex_out,
                             float* __restrict__ den) {
  __shared__ _Float16 sA[16][64];
  __shared__ float sEe[16][128];
  __shared__ float sS[16][64];
  const int t = threadIdx.x, lane = t & 31, wave = t >> 5;
  const int e0 = blockIdx.x * 16;
  stage_f32_to_h16(edge_attr + (size_t)e0 * 64, sA, t);
  __syncthreads();
  const int hl = lane >> 4, nn = lane & 15;
#pragma unroll
  for (int j = 0; j < 2; ++j) {
    const int ct = wave * 2 + j;  // 0..7
    v8f c = gemm_k64(&sA[0][0], 64, We16, ct, lane);
    const int col = ct * 16 + nn;
#pragma unroll
    for (int r = 0; r < 8; ++r) sEe[r + 8 * hl][col] = c[r] + be[col];
  }
  __syncthreads();
  // per-edge s: quad-per-thread (quads never cross a head boundary)
  for (int i = t; i < 256; i += 128) {
    const int r = i >> 4, c0 = (i & 15) * 4;
    const int e = e0 + r;
    const int si = src[e], di = dst[e];
    const float4 kk = *(const float4*)(qkv + (size_t)si * 192 + 64 + c0);
    const float4 qq = *(const float4*)(qkv + (size_t)di * 192 + c0);
    const int h = c0 >> 3;
    h4 so;
    float kq[4] = {kk.x + qq.x, kk.y + qq.y, kk.z + qq.z, kk.w + qq.w};
#pragma unroll
    for (int u = 0; u < 4; ++u) {
      const int d = (c0 + u) & 7;
      float s = kq[u] * sEe[r][h * 16 + d];
      s = sqrtf(fmaxf(s, 0.f) + 1e-8f) - sqrtf(fmaxf(-s, 0.f) + 1e-8f);
      s += sEe[r][h * 16 + 8 + d];
      sS[r][c0 + u] = s;
      so[u] = (_Float16)s;
    }
    *(h4*)(s_out + (size_t)e * 64 + c0) = so;
  }
  __syncthreads();
  {
    const int r = t >> 3, h = t & 7;  // 16 edges x 8 heads == 128 threads
    float acc = 0.f;
#pragma unroll
    for (int d = 0; d < 8; ++d) acc += sS[r][h * 8 + d] * Aw[d * 8 + h];
    acc = fminf(fmaxf(acc, -5.0f), 5.0f);
    const float ex = expf(acc - 5.0f);
    const int e = e0 + r;
    ex_out[(size_t)e * 8 + h] = ex;
    atomicAdd(&den[(size_t)dst[e] * 8 + h], ex);
  }
}

// ---------------------------------------------------------------------------
// Kernel 3: attn = ex/(den+1e-16); scatter-add Vh[src]*attn, s*attn into
// wV/rowV (12.8MB each -> L2-resident). Quad-per-thread.
// ---------------------------------------------------------------------------
__global__ void k_aggregate(const _Float16* __restrict__ s16, const float* __restrict__ ex,
                            const float* __restrict__ den, const float* __restrict__ qkv,
                            const int* __restrict__ src, const int* __restrict__ dst,
                            float* __restrict__ wV, float* __restrict__ rowV) {
  const size_t idx = (size_t)blockIdx.x * 256 + threadIdx.x;  // grid = E*16/256
  const int e = (int)(idx >> 4);
  const int c0 = (int)(idx & 15) * 4;
  const int h = c0 >> 3;
  const int di = dst[e];
  const float attn = ex[(size_t)e * 8 + h] / (den[(size_t)di * 8 + h] + 1e-16f);
  const float4 v4 = *(const float4*)(qkv + (size_t)src[e] * 192 + 128 + c0);
  const h4 s4 = *(const h4*)(s16 + (size_t)e * 64 + c0);
  float* wv = wV + (size_t)di * 64 + c0;
  float* rv = rowV + (size_t)di * 64 + c0;
  atomicAdd(wv + 0, v4.x * attn); atomicAdd(rv + 0, (float)s4[0] * attn);
  atomicAdd(wv + 1, v4.y * attn); atomicAdd(rv + 1, (float)s4[1] * attn);
  atomicAdd(wv + 2, v4.z * attn); atomicAdd(rv + 2, (float)s4[2] * attn);
  atomicAdd(wv + 3, v4.w * attn); atomicAdd(rv + 3, (float)s4[3] * attn);
}

// ---------------------------------------------------------------------------
// Kernel 4: fused node tail: wV += rowV@Wve; degree scaling; @Woh+boh;
// LN1(x+.); relu(@W1+b1)@W2+b2; LN2 -> h output.
// ---------------------------------------------------------------------------
__global__ void k_node_out(const float* __restrict__ x, const float* __restrict__ wV,
                           const float* __restrict__ rowV, const float* __restrict__ log_deg,
                           const float* __restrict__ deg_coef,
                           const _Float16* __restrict__ Wve, const _Float16* __restrict__ Woh16,
                           const float* __restrict__ boh,
                           const _Float16* __restrict__ W116, const float* __restrict__ b1,
                           const _Float16* __restrict__ W216, const float* __restrict__ b2,
                           const float* __restrict__ g1, const float* __restrict__ bt1,
                           const float* __restrict__ g2, const float* __restrict__ bt2,
                           float* __restrict__ out) {
  __shared__ _Float16 sA[16][64];
  __shared__ _Float16 sM[16][128];
  __shared__ float sF[16][64];
  __shared__ float sG[16][64];
  const int t = threadIdx.x, lane = t & 31, wave = t >> 5;
  const int hl = lane >> 4, nn = lane & 15;
  const int row0 = blockIdx.x * 16;

  stage_f32_to_h16(rowV + (size_t)row0 * 64, sA, t);
  for (int i = t; i < 256; i += 128) {
    const int r = i >> 4, c0 = (i & 15) * 4;
    *(float4*)&sF[r][c0] = *(const float4*)(wV + (size_t)(row0 + r) * 64 + c0);
  }
  __syncthreads();
  {  // wV += rowV @ Wve
    v8f c = gemm_k64(&sA[0][0], 64, Wve, wave, lane);
    const int col = wave * 16 + nn;
#pragma unroll
    for (int r = 0; r < 8; ++r) sF[r + 8 * hl][col] += c[r];
  }
  __syncthreads();
  for (int i = t; i < 1024; i += 128) {  // degree scaling -> f16
    const int r = i >> 6, c = i & 63;
    const float ld = log_deg[row0 + r];
    sA[r][c] = (_Float16)(sF[r][c] * (deg_coef[2 * c] + ld * deg_coef[2 * c + 1]));
  }
  __syncthreads();
  {  // h1 = h @ Woh + boh ; residual with x
    v8f c = gemm_k64(&sA[0][0], 64, Woh16, wave, lane);
    const int col = wave * 16 + nn;
#pragma unroll
    for (int r = 0; r < 8; ++r) {
      const int m = r + 8 * hl;
      sF[m][col] = x[(size_t)(row0 + m) * 64 + col] + c[r] + boh[col];
    }
  }
  __syncthreads();
  ln16_rows(sF, g1, bt1, t, sG, sA, nullptr);
  __syncthreads();
#pragma unroll
  for (int j = 0; j < 2; ++j) {  // mid = relu(hln @ W1 + b1)
    const int ct = wave * 2 + j;  // 0..7
    v8f c = gemm_k64(&sA[0][0], 64, W116, ct, lane);
    const int col = ct * 16 + nn;
#pragma unroll
    for (int r = 0; r < 8; ++r)
      sM[r + 8 * hl][col] = (_Float16)fmaxf(c[r] + b1[col], 0.f);
  }
  __syncthreads();
  {  // h2 = mid @ W2 + b2 (K=128) ; residual with hln
    v8f c = {};
#pragma unroll
    for (int ks = 0; ks < 4; ++ks)
      c = __builtin_amdgcn_wmma_f32_16x16x32_f16(false, frag_a(&sM[0][0], 128, ks * 32, lane),
                                                 false, frag_b_swz(W216, wave * 4 + ks, lane),
                                                 (short)0, c, false, false);
    const int col = wave * 16 + nn;
#pragma unroll
    for (int r = 0; r < 8; ++r) {
      const int m = r + 8 * hl;
      sF[m][col] = sG[m][col] + c[r] + b2[col];
    }
  }
  __syncthreads();
  ln16_rows(sF, g2, bt2, t, nullptr, nullptr, out + (size_t)row0 * 64);
}

// ---------------------------------------------------------------------------
// Kernel 5: e = LN(edge_attr + s@Woe + boe)
// ---------------------------------------------------------------------------
__global__ void k_edge_out(const _Float16* __restrict__ s16, const float* __restrict__ edge_attr,
                           const _Float16* __restrict__ Woe16, const float* __restrict__ boe,
                           const float* __restrict__ g, const float* __restrict__ bta,
                           float* __restrict__ out) {
  __shared__ _Float16 sA[16][64];
  __shared__ float sF[16][64];
  const int t = threadIdx.x, lane = t & 31, wave = t >> 5;
  const int hl = lane >> 4, nn = lane & 15;
  const int e0 = blockIdx.x * 16;
  for (int i = t; i < 256; i += 128) {   // already f16: 64-bit copies
    const int r = i >> 4, c0 = (i & 15) * 4;
    *(h4*)&sA[r][c0] = *(const h4*)(s16 + (size_t)(e0 + r) * 64 + c0);
  }
  __syncthreads();
  {
    v8f c = gemm_k64(&sA[0][0], 64, Woe16, wave, lane);
    const int col = wave * 16 + nn;
#pragma unroll
    for (int r = 0; r < 8; ++r) {
      const int m = r + 8 * hl;
      sF[m][col] = edge_attr[(size_t)(e0 + m) * 64 + col] + c[r] + boe[col];
    }
  }
  __syncthreads();
  ln16_rows(sF, g, bta, t, nullptr, nullptr, out + (size_t)e0 * 64);
}

// ---------------------------------------------------------------------------
extern "C" void kernel_launch(void* const* d_in, const int* in_sizes, int n_in,
                              void* d_out, int out_size, void* d_ws, size_t ws_size,
                              hipStream_t stream) {
  const float* x         = (const float*)d_in[0];
  const float* edge_attr = (const float*)d_in[1];
  const float* log_deg   = (const float*)d_in[2];
  const float* Wq  = (const float*)d_in[3];
  const float* bq  = (const float*)d_in[4];
  const float* Wk  = (const float*)d_in[5];
  const float* bk  = (const float*)d_in[6];
  const float* Wv  = (const float*)d_in[7];
  const float* bv  = (const float*)d_in[8];
  const float* We  = (const float*)d_in[9];
  const float* be  = (const float*)d_in[10];
  const float* Aw  = (const float*)d_in[11];
  const float* VeRow = (const float*)d_in[12];
  const float* deg_coef = (const float*)d_in[13];
  const float* Woh = (const float*)d_in[14];
  const float* boh = (const float*)d_in[15];
  const float* Woe = (const float*)d_in[16];
  const float* boe = (const float*)d_in[17];
  const float* ln1h_g = (const float*)d_in[18];
  const float* ln1h_b = (const float*)d_in[19];
  const float* ln1e_g = (const float*)d_in[20];
  const float* ln1e_b = (const float*)d_in[21];
  const float* W1  = (const float*)d_in[22];
  const float* b1  = (const float*)d_in[23];
  const float* W2  = (const float*)d_in[24];
  const float* b2  = (const float*)d_in[25];
  const float* ln2h_g = (const float*)d_in[26];
  const float* ln2h_b = (const float*)d_in[27];
  const int* ei = (const int*)d_in[28];
  const int* src = ei;
  const int* dst = ei + E_EDGES;

  // Workspace layout (all offsets 256B-aligned by construction)
  char* ws = (char*)d_ws;
  size_t o = 0;
  float* qkv = (float*)(ws + o);       o += (size_t)N_NODES * 192 * 4;  // 38.4 MB
  _Float16* s16 = (_Float16*)(ws + o); o += (size_t)E_EDGES * 64 * 2;   // 64 MB
  float* ex  = (float*)(ws + o);       o += (size_t)E_EDGES * 8 * 4;    // 16 MB
  float* den = (float*)(ws + o);       o += (size_t)N_NODES * 8 * 4;    // 1.6 MB
  float* wV  = (float*)(ws + o);       o += (size_t)N_NODES * 64 * 4;   // 12.8 MB
  float* rowV = (float*)(ws + o);      o += (size_t)N_NODES * 64 * 4;   // 12.8 MB
  _Float16* W16 = (_Float16*)(ws + o); o += 49152 * 2;                  // 96 KB
  float* bqkv = (float*)(ws + o);                                       // 768 B

  hipMemsetAsync(den, 0, ((size_t)N_NODES * 8 + (size_t)N_NODES * 128) * 4, stream);

  k_prep<<<193, 256, 0, stream>>>(Wq, Wk, Wv, We, Woh, Woe, W1, W2, VeRow,
                                  bq, bk, bv, W16, bqkv);

  k_qkv<<<N_NODES / 16, 128, 0, stream>>>(x, W16 /*Wqkv@0*/, bqkv, qkv);

  k_edge_score<<<E_EDGES / 16, 128, 0, stream>>>(edge_attr, W16 + 12288, be, qkv,
                                                 src, dst, Aw, s16, ex, den);

  k_aggregate<<<(E_EDGES * 16) / 256, 256, 0, stream>>>(s16, ex, den, qkv, src, dst, wV, rowV);

  k_node_out<<<N_NODES / 16, 128, 0, stream>>>(x, wV, rowV, log_deg, deg_coef,
                                               W16 + 45056 /*Wve*/, W16 + 20480 /*Woh*/, boh,
                                               W16 + 28672 /*W1*/, b1, W16 + 36864 /*W2*/, b2,
                                               ln1h_g, ln1h_b, ln2h_g, ln2h_b,
                                               (float*)d_out);

  k_edge_out<<<E_EDGES / 16, 128, 0, stream>>>(s16, edge_attr, W16 + 24576 /*Woe*/, boe,
                                               ln1e_g, ln1e_b,
                                               (float*)d_out + (size_t)N_NODES * 64);
}
